// NSDEModel_54168127537634
// MI455X (gfx1250) — compile-verified, matrix operand order checked by reading, not run on hardware
//
#include <hip/hip_runtime.h>
#include <hip/hip_bf16.h>

typedef __attribute__((ext_vector_type(16))) _Float16 v16h;
typedef __attribute__((ext_vector_type(8)))  _Float16 v8h;
typedef __attribute__((ext_vector_type(8)))  float    v8f;
typedef __attribute__((ext_vector_type(4)))  float    v4f;

#define B_ROWS 4096
#define L_LEN  256
#define F_DIM  32
#define M_DIM  16
#define H_DIM  64
#define STEPS  20
#define DT_C       0.05f
#define SQRT_DT_C  0.22360679774997896f

#define WAVES_PER_BLOCK 2
#define BLOCK_THREADS   (WAVES_PER_BLOCK * 32)
#define ROWS_PER_WAVE   16
#define ROWS_PER_BLOCK  (WAVES_PER_BLOCK * ROWS_PER_WAVE)
#define GRID_BLOCKS     (B_ROWS / ROWS_PER_BLOCK)

#define ACT_STRIDE 136   // halfs per activation row (rows land on distinct LDS banks)
#define H_STRIDE   68    // floats per h / noise row (banks: +4 dwords per row)

// ---- LDS map (bytes). All offsets multiples of 32 for v16h alignment. ----
#define OFF_W1   0                        // 4 ktiles x 8 ntiles x 512 halfs
#define OFF_W2   32768                    // 4 x 8
#define OFF_W3   65536                    // 4 x 4
#define OFF_WD1  81920                    // 2 x 4
#define OFF_WD2  90112                    // 2 x 4
#define OFF_WR1  98304                    // 2 x 2
#define OFF_WR2  102400                   // 1 x 1
#define OFF_BIAS 103424                   // 560 floats
#define OFF_WAVE 105664
// per-wave region: inp | zA | zB | h | noise
#define WOFF_INP  0
#define WOFF_ZA   4352
#define WOFF_ZB   8704
#define WOFF_HB   13056
#define WOFF_NB   17408                   // 16 rows x 68 floats (272 B/row)
#define WAVE_BYTES 21760
#define SMEM_BYTES (OFF_WAVE + WAVES_PER_BLOCK * WAVE_BYTES)   // 149184 B < 320 KB

// bias sub-offsets (floats)
#define BO_B1   0
#define BO_B2   128
#define BO_B3   256
#define BO_BD1  320
#define BO_BD2  384
#define BO_SCALE 448
#define BO_BR1  512
#define BO_BR2  544

// Stage a KxN f32 weight matrix into LDS as f16 WMMA B-fragments, zero-padded.
// B-fragment lane layout (16-bit, 32x16): lane c: N=c, K={k0..k0+7, k0+16..k0+23};
// lane 16+c: N=c, K={k0+8..k0+15, k0+24..k0+31}. 16 halfs contiguous per lane.
__device__ inline void stage_wfrag(const float* __restrict__ W, int K, int N,
                                   int ktiles, int ntiles, _Float16* dst,
                                   int tid, int nth) {
  int total = ktiles * ntiles * 32;
  for (int s = tid; s < total; s += nth) {
    int lane = s & 31;
    int tile = s >> 5;
    int nt = tile % ntiles;
    int kt = tile / ntiles;
    int n  = nt * 16 + (lane & 15);
    int kb = kt * 32 + ((lane & 16) ? 8 : 0);
    _Float16* out = dst + (size_t)tile * 512 + lane * 16;
#pragma unroll
    for (int j = 0; j < 16; ++j) {
      int k = kb + (j & 7) + ((j & 8) ? 16 : 0);
      float v = (k < K && n < N) ? W[(size_t)k * N + n] : 0.0f;
      out[j] = (_Float16)v;
    }
  }
}

// A-fragment (16x32 f16) from row-major LDS: two contiguous 16B chunks per lane.
__device__ inline v16h load_afrag(const _Float16* act, int k0, int lane) {
  int m  = lane & 15;
  int kb = k0 + ((lane & 16) ? 8 : 0);
  const _Float16* p = act + m * ACT_STRIDE + kb;
  v8h lo = *(const v8h*)p;
  v8h hi = *(const v8h*)(p + 16);
  return __builtin_shufflevector(lo, hi, 0,1,2,3,4,5,6,7,8,9,10,11,12,13,14,15);
}

template<int KT>
__device__ inline void load_afrags(const _Float16* act, int lane, v16h (&a)[KT]) {
#pragma unroll
  for (int kt = 0; kt < KT; ++kt) a[kt] = load_afrag(act, kt * 32, lane);
}

__device__ inline v8f wmma16(v16h a, v16h b, v8f c) {
  return __builtin_amdgcn_wmma_f32_16x16x32_f16(false, a, false, b, (short)0, c, false, false);
}

// 4 N-tiles sharing register-resident A-fragments: 4 independent WMMA chains.
template<int KT>
__device__ inline void gemm_quad(const v16h (&a)[KT], const _Float16* wf,
                                 int ntiles, int nt0, int lane,
                                 v8f& c0, v8f& c1, v8f& c2, v8f& c3) {
  c0 = {}; c1 = {}; c2 = {}; c3 = {};
#pragma unroll
  for (int kt = 0; kt < KT; ++kt) {
    const _Float16* p = wf + (size_t)(kt * ntiles + nt0) * 512 + lane * 16;
    v16h b0 = *(const v16h*)(p);
    v16h b1 = *(const v16h*)(p + 512);
    v16h b2 = *(const v16h*)(p + 1024);
    v16h b3 = *(const v16h*)(p + 1536);
    c0 = wmma16(a[kt], b0, c0);
    c1 = wmma16(a[kt], b1, c1);
    c2 = wmma16(a[kt], b2, c2);
    c3 = wmma16(a[kt], b3, c3);
  }
}

template<int KT>
__device__ inline void gemm_pair(const v16h (&a)[KT], const _Float16* wf,
                                 int ntiles, int nt0, int lane, v8f& c0, v8f& c1) {
  c0 = {}; c1 = {};
#pragma unroll
  for (int kt = 0; kt < KT; ++kt) {
    const _Float16* p = wf + (size_t)(kt * ntiles + nt0) * 512 + lane * 16;
    v16h b0 = *(const v16h*)(p);
    v16h b1 = *(const v16h*)(p + 512);
    c0 = wmma16(a[kt], b0, c0);
    c1 = wmma16(a[kt], b1, c1);
  }
}

// C-fragment (lane<16: M=v,N=lane ; lane>=16: M=v+8,N=lane-16) + bias, ReLU, f16 row-major.
__device__ inline void store_act_relu(v8f c, const float* bias, int nt, _Float16* dst, int lane) {
  int n  = nt * 16 + (lane & 15);
  float bv = bias[n];
  int mb = (lane & 16) ? 8 : 0;
#pragma unroll
  for (int v = 0; v < 8; ++v) {
    float x = c[v] + bv;
    dst[(mb + v) * ACT_STRIDE + n] = (_Float16)(x > 0.0f ? x : 0.0f);
  }
}

__global__ __launch_bounds__(BLOCK_THREADS)
void nsde_kernel(const float* __restrict__ x_path, const float* __restrict__ macro_path,
                 const float* __restrict__ t_span, const float* __restrict__ noise,
                 const float* __restrict__ W1,  const float* __restrict__ b1,
                 const float* __restrict__ W2,  const float* __restrict__ b2,
                 const float* __restrict__ W3,  const float* __restrict__ b3,
                 const float* __restrict__ Wd1, const float* __restrict__ bd1,
                 const float* __restrict__ Wd2, const float* __restrict__ bd2,
                 const float* __restrict__ scale, const float* __restrict__ Wr1,
                 const float* __restrict__ br1, const float* __restrict__ Wr2,
                 const float* __restrict__ br2, float* __restrict__ out) {
  extern __shared__ char smem[];
  const int tid  = threadIdx.x;
  const int lane = tid & 31;
  const int wave = tid >> 5;
  const int r0   = blockIdx.x * ROWS_PER_BLOCK + wave * ROWS_PER_WAVE;

  _Float16* wfW1  = (_Float16*)(smem + OFF_W1);
  _Float16* wfW2  = (_Float16*)(smem + OFF_W2);
  _Float16* wfW3  = (_Float16*)(smem + OFF_W3);
  _Float16* wfWd1 = (_Float16*)(smem + OFF_WD1);
  _Float16* wfWd2 = (_Float16*)(smem + OFF_WD2);
  _Float16* wfWr1 = (_Float16*)(smem + OFF_WR1);
  _Float16* wfWr2 = (_Float16*)(smem + OFF_WR2);
  float*    bl    = (float*)(smem + OFF_BIAS);
  char*     wbp   = smem + OFF_WAVE + (size_t)wave * WAVE_BYTES;
  _Float16* inp   = (_Float16*)(wbp + WOFF_INP);
  _Float16* zA    = (_Float16*)(wbp + WOFF_ZA);
  _Float16* zB    = (_Float16*)(wbp + WOFF_ZB);
  float*    hb    = (float*)(wbp + WOFF_HB);
  const float* nb = (const float*)(wbp + WOFF_NB);
  const uint32_t nbuf_lds = (uint32_t)(OFF_WAVE + wave * WAVE_BYTES + WOFF_NB);

  // ---- one-time staging: weights -> f16 B-fragments, biases -> LDS ----
  stage_wfrag(W1,  112, 128, 4, 8, wfW1,  tid, BLOCK_THREADS);
  stage_wfrag(W2,  128, 128, 4, 8, wfW2,  tid, BLOCK_THREADS);
  stage_wfrag(W3,  128,  64, 4, 4, wfW3,  tid, BLOCK_THREADS);
  stage_wfrag(Wd1,  64,  64, 2, 4, wfWd1, tid, BLOCK_THREADS);
  stage_wfrag(Wd2,  64,  64, 2, 4, wfWd2, tid, BLOCK_THREADS);
  stage_wfrag(Wr1,  64,  32, 2, 2, wfWr1, tid, BLOCK_THREADS);
  stage_wfrag(Wr2,  32,   2, 1, 1, wfWr2, tid, BLOCK_THREADS);
  for (int i = tid; i < 128; i += BLOCK_THREADS) { bl[BO_B1 + i] = b1[i]; bl[BO_B2 + i] = b2[i]; }
  for (int i = tid; i < 64;  i += BLOCK_THREADS) {
    bl[BO_B3 + i] = b3[i]; bl[BO_BD1 + i] = bd1[i];
    bl[BO_BD2 + i] = bd2[i]; bl[BO_SCALE + i] = scale[i];
  }
  for (int i = tid; i < 32; i += BLOCK_THREADS) bl[BO_BR1 + i] = br1[i];
  for (int i = tid; i < 16; i += BLOCK_THREADS) bl[BO_BR2 + i] = (i < 2) ? br2[i] : 0.0f;

  // init h = 0 and constant zero padding of inp cols 112..127 (wave-private)
  for (int i = lane; i < ROWS_PER_WAVE * H_STRIDE; i += 32) hb[i] = 0.0f;
  if (lane < 16)
    for (int m2 = 0; m2 < 16; ++m2) inp[m2 * ACT_STRIDE + 112 + lane] = (_Float16)0.0f;

  __syncthreads();   // weights/biases visible to both waves; only block-wide barrier needed

  const float tN = t_span[STEPS];
  const int m  = lane & 15;
  const int mb = (lane & 16) ? 8 : 0;

  for (int s = 0; s < STEPS; ++s) {
    // ---- async-copy this step's noise slice (16 rows x 256B) into LDS (ASYNCcnt) ----
    {
      const char* gb = (const char*)noise + ((size_t)s * B_ROWS + r0) * H_DIM * 4;
#pragma unroll
      for (int i = 0; i < 8; ++i) {
        int j = i * 32 + lane;                                   // 16B segment index
        uint32_t goff = (uint32_t)j * 16u;
        uint32_t loff = nbuf_lds + (uint32_t)(j >> 4) * 272u + (uint32_t)(j & 15) * 16u;
        asm volatile("global_load_async_to_lds_b128 %0, %1, %2"
                     :: "v"(loff), "v"(goff), "s"(gb) : "memory");
      }
    }

    // time index, matching jnp truncation + clip
    float tn = t_span[s] / tN;
    int ix = (int)(tn * (float)(L_LEN - 1));
    ix = ix < 0 ? 0 : (ix > L_LEN - 1 ? L_LEN - 1 : ix);

    // ---- build inp = [h | x_t | m_t | 0] as f16 (vectorized 16B loads) ----
    if (lane < 16) {
      const v4f* hr = (const v4f*)(hb + m * H_STRIDE);
      _Float16* dr = inp + m * ACT_STRIDE;
#pragma unroll
      for (int c4 = 0; c4 < 16; ++c4) {
        v4f h4 = hr[c4];
#pragma unroll
        for (int j = 0; j < 4; ++j) dr[c4 * 4 + j] = (_Float16)h4[j];
      }
    } else {
      const v4f* xr = (const v4f*)(x_path     + ((size_t)(r0 + m) * L_LEN + ix) * F_DIM);
      const v4f* mr = (const v4f*)(macro_path + ((size_t)(r0 + m) * L_LEN + ix) * M_DIM);
      _Float16* dr = inp + m * ACT_STRIDE;
#pragma unroll
      for (int c4 = 0; c4 < 8; ++c4) {
        v4f x4 = xr[c4];
#pragma unroll
        for (int j = 0; j < 4; ++j) dr[64 + c4 * 4 + j] = (_Float16)x4[j];
      }
#pragma unroll
      for (int c4 = 0; c4 < 4; ++c4) {
        v4f m4 = mr[c4];
#pragma unroll
        for (int j = 0; j < 4; ++j) dr[96 + c4 * 4 + j] = (_Float16)m4[j];
      }
    }

    // prefetch next step's x/macro rows (global_prefetch_b8)
    if (s + 1 < STEPS) {
      float tn2 = t_span[s + 1] / tN;
      int ix2 = (int)(tn2 * (float)(L_LEN - 1));
      ix2 = ix2 < 0 ? 0 : (ix2 > L_LEN - 1 ? L_LEN - 1 : ix2);
      const char* px = (const char*)(x_path + ((size_t)(r0 + m) * L_LEN + ix2) * F_DIM);
      __builtin_prefetch(px, 0, 0);
      __builtin_prefetch(px + 64, 0, 0);
      __builtin_prefetch((const char*)(macro_path + ((size_t)(r0 + m) * L_LEN + ix2) * M_DIM), 0, 0);
    }

    v8f c0, c1, c2, c3;

    // drift layer 1: zA = relu(inp @ W1 + b1)   K=112(pad128), N=128
    v16h aIn[4];
    load_afrags<4>(inp, lane, aIn);
    gemm_quad<4>(aIn, wfW1, 8, 0, lane, c0, c1, c2, c3);
    store_act_relu(c0, bl + BO_B1, 0, zA, lane);
    store_act_relu(c1, bl + BO_B1, 1, zA, lane);
    store_act_relu(c2, bl + BO_B1, 2, zA, lane);
    store_act_relu(c3, bl + BO_B1, 3, zA, lane);
    gemm_quad<4>(aIn, wfW1, 8, 4, lane, c0, c1, c2, c3);
    store_act_relu(c0, bl + BO_B1, 4, zA, lane);
    store_act_relu(c1, bl + BO_B1, 5, zA, lane);
    store_act_relu(c2, bl + BO_B1, 6, zA, lane);
    store_act_relu(c3, bl + BO_B1, 7, zA, lane);
    v16h hA[2] = {aIn[0], aIn[1]};   // h16 fragments, reused by diffusion layer 1

    // drift layer 2: zB = relu(zA @ W2 + b2)
    v16h aZ[4];
    load_afrags<4>(zA, lane, aZ);
    gemm_quad<4>(aZ, wfW2, 8, 0, lane, c0, c1, c2, c3);
    store_act_relu(c0, bl + BO_B2, 0, zB, lane);
    store_act_relu(c1, bl + BO_B2, 1, zB, lane);
    store_act_relu(c2, bl + BO_B2, 2, zB, lane);
    store_act_relu(c3, bl + BO_B2, 3, zB, lane);
    gemm_quad<4>(aZ, wfW2, 8, 4, lane, c0, c1, c2, c3);
    store_act_relu(c0, bl + BO_B2, 4, zB, lane);
    store_act_relu(c1, bl + BO_B2, 5, zB, lane);
    store_act_relu(c2, bl + BO_B2, 6, zB, lane);
    store_act_relu(c3, bl + BO_B2, 7, zB, lane);

    // drift layer 3 in registers: drift = zB @ W3 (+b3 fused later)
    v16h aB[4];
    load_afrags<4>(zB, lane, aB);
    v8f d0, d1, d2, d3;
    gemm_quad<4>(aB, wfW3, 4, 0, lane, d0, d1, d2, d3);

    // diffusion layer 1: zA = relu(h16 @ Wd1 + bd1)  (A-fragments cached in regs)
    gemm_quad<2>(hA, wfWd1, 4, 0, lane, c0, c1, c2, c3);
    store_act_relu(c0, bl + BO_BD1, 0, zA, lane);
    store_act_relu(c1, bl + BO_BD1, 1, zA, lane);
    store_act_relu(c2, bl + BO_BD1, 2, zA, lane);
    store_act_relu(c3, bl + BO_BD1, 3, zA, lane);

    // diffusion layer 2
    v16h aD[2];
    load_afrags<2>(zA, lane, aD);
    gemm_quad<2>(aD, wfWd2, 4, 0, lane, c0, c1, c2, c3);

    // ---- fused Euler-Maruyama update: h += drift*DT + scale*sigmoid(.)*sqrt(DT)*z ----
    asm volatile("s_wait_asynccnt 0" ::: "memory");   // noise slice now in LDS
    v8f dsv[4] = {d0, d1, d2, d3};
    v8f csv[4] = {c0, c1, c2, c3};
#pragma unroll
    for (int nt = 0; nt < 4; ++nt) {
      int n = nt * 16 + m;
      float bv  = bl[BO_BD2 + n];
      float sc  = bl[BO_SCALE + n];
      float b3v = bl[BO_B3 + n];
#pragma unroll
      for (int v = 0; v < 8; ++v) {
        float g = sc / (1.0f + __expf(-(csv[nt][v] + bv)));
        float z = nb[(mb + v) * H_STRIDE + n];
        hb[(mb + v) * H_STRIDE + n] += (dsv[nt][v] + b3v) * DT_C + g * SQRT_DT_C * z;
      }
    }
  }

  // ---- readout: out = relu(h @ Wr1 + br1) @ Wr2 + br2 ----
  {
    int cbase = (lane < 16) ? 0 : 32;
    const v4f* hr = (const v4f*)(hb + m * H_STRIDE + cbase);
    _Float16* dr = inp + m * ACT_STRIDE + cbase;
#pragma unroll
    for (int c4 = 0; c4 < 8; ++c4) {
      v4f h4 = hr[c4];
#pragma unroll
      for (int j = 0; j < 4; ++j) dr[c4 * 4 + j] = (_Float16)h4[j];
    }
  }
  {
    v16h aH[2];
    load_afrags<2>(inp, lane, aH);
    v8f c0, c1;
    gemm_pair<2>(aH, wfWr1, 2, 0, lane, c0, c1);
    store_act_relu(c0, bl + BO_BR1, 0, zA, lane);
    store_act_relu(c1, bl + BO_BR1, 1, zA, lane);
  }
  {
    v8f c = {};
    v16h a = load_afrag(zA, 0, lane);                 // K = 32: cols 0..31 of zA
    v16h b = *(const v16h*)(wfWr2 + lane * 16);
    c = __builtin_amdgcn_wmma_f32_16x16x32_f16(false, a, false, b, (short)0, c, false, false);
    if (m < 2) {                                      // only output cols 0,1 are real
      float bv = bl[BO_BR2 + m];
#pragma unroll
      for (int v = 0; v < 8; ++v)
        out[(size_t)m * B_ROWS + (r0 + mb + v)] = c[v] + bv;   // out[:,0] then out[:,1]
    }
  }
}

extern "C" void kernel_launch(void* const* d_in, const int* in_sizes, int n_in,
                              void* d_out, int out_size, void* d_ws, size_t ws_size,
                              hipStream_t stream) {
  (void)in_sizes; (void)n_in; (void)out_size; (void)d_ws; (void)ws_size;
  const float* x_path     = (const float*)d_in[0];
  const float* macro_path = (const float*)d_in[1];
  const float* t_span     = (const float*)d_in[2];
  const float* noise      = (const float*)d_in[3];
  const float* W1  = (const float*)d_in[4];
  const float* b1  = (const float*)d_in[5];
  const float* W2  = (const float*)d_in[6];
  const float* b2  = (const float*)d_in[7];
  const float* W3  = (const float*)d_in[8];
  const float* b3  = (const float*)d_in[9];
  const float* Wd1 = (const float*)d_in[10];
  const float* bd1 = (const float*)d_in[11];
  const float* Wd2 = (const float*)d_in[12];
  const float* bd2 = (const float*)d_in[13];
  const float* scl = (const float*)d_in[14];
  const float* Wr1 = (const float*)d_in[15];
  const float* br1 = (const float*)d_in[16];
  const float* Wr2 = (const float*)d_in[17];
  const float* br2 = (const float*)d_in[18];
  float* out = (float*)d_out;

  // opt in to >64KB dynamic LDS (CDNA5: up to 320KB per workgroup)
  hipFuncSetAttribute((const void*)nsde_kernel,
                      hipFuncAttributeMaxDynamicSharedMemorySize, SMEM_BYTES);

  nsde_kernel<<<GRID_BLOCKS, BLOCK_THREADS, SMEM_BYTES, stream>>>(
      x_path, macro_path, t_span, noise,
      W1, b1, W2, b2, W3, b3, Wd1, bd1, Wd2, bd2, scl, Wr1, br1, Wr2, br2, out);
}
